// AQD_GCN_48567490183789
// MI455X (gfx1250) — matrix operand order repaired and checked by reading, not compiled, not run on hardware
//
#include <hip/hip_runtime.h>

// ---------------------------------------------------------------------------
// AQD_GCN forward for MI455X (gfx1250, wave32, WMMA).
// One bf16 WMMA GEMM (fp32 accumulate) drives all matmuls.
// High-reuse operands (adj, Fadj, Fadj^T) pre-converted to bf16 once.
// Staging uses vectorized b128 global loads + wide LDS stores on the fast
// path (full tiles); scalar guarded path only for K=1/2 edge tiles.
// ---------------------------------------------------------------------------

typedef __bf16 bf16_t;
typedef __attribute__((ext_vector_type(16))) __bf16 v16bf;
typedef __attribute__((ext_vector_type(8)))  __bf16 v8bf;
typedef __attribute__((ext_vector_type(4)))  __bf16 v4bf;
typedef __attribute__((ext_vector_type(8)))  float  v8f;

#define GEMM_TM 128
#define GEMM_TN 128
#define GEMM_TK 32
#define ASTR    40    // padded LDS stride (halves): 80B rows -> 16B aligned, bank-spread
#define BSTR    40

union frag16 { v16bf v; v8bf h[2]; };

// C[M,N] (+)= A[M,K] * B[K,N] (+ bias[N]).  Row-major.
// 8 waves: 4 row-blocks x 2 col-blocks; each wave -> 32x64 (8 WMMAs / K-step).
template <typename TA>
__global__ __launch_bounds__(256) void aqd_gemm_wmma(
    const TA* __restrict__ Ag, int lda,
    const float* __restrict__ Bg, int ldb,
    float* __restrict__ Cg, int ldc,
    int Mrows, int Ncols, int Kdim,
    const float* __restrict__ bias, int accumulate)
{
    __shared__ __align__(16) bf16_t As[GEMM_TM * ASTR];   // [row][k]
    __shared__ __align__(16) bf16_t Bs[GEMM_TN * BSTR];   // [col][k]  (col-major tile)

    const int tid  = threadIdx.x;
    const int wave = tid >> 5;
    const int lane = tid & 31;
    const int lm   = lane & 15;     // index within 16
    const int lg   = lane >> 4;     // lane group (0/1)
    const int wr   = wave & 3;      // 4 row blocks of 32
    const int wc   = wave >> 2;     // 2 col blocks of 64
    const int row0 = blockIdx.y * GEMM_TM;
    const int col0 = blockIdx.x * GEMM_TN;

    v8f acc[2][4] = {};

    const bool aFull = (row0 + GEMM_TM <= Mrows) && ((lda & 3) == 0);
    const bool bFull = (col0 + GEMM_TN <= Ncols) && ((ldb & 3) == 0);

    for (int k0 = 0; k0 < Kdim; k0 += GEMM_TK) {
        const bool kFull = (k0 + GEMM_TK <= Kdim);

        // ---- stage A tile (-> bf16) ----
        if (aFull && kFull) {
            // 512 chunks of 8 elements; 2 per thread; b128 loads, b128 LDS stores
            #pragma unroll
            for (int t = 0; t < 2; ++t) {
                int i  = tid + t * 256;          // 0..511
                int r  = i >> 2;                 // 4 chunks per row
                int kc = (i & 3) * 8;
                const TA* src = Ag + (size_t)(row0 + r) * lda + k0 + kc;
                v8bf out;
                if constexpr (sizeof(TA) == 2) {
                    out = *(const v8bf*)src;
                } else {
                    float4 x = ((const float4*)src)[0];
                    float4 y = ((const float4*)src)[1];
                    out[0] = (bf16_t)x.x; out[1] = (bf16_t)x.y;
                    out[2] = (bf16_t)x.z; out[3] = (bf16_t)x.w;
                    out[4] = (bf16_t)y.x; out[5] = (bf16_t)y.y;
                    out[6] = (bf16_t)y.z; out[7] = (bf16_t)y.w;
                }
                *(v8bf*)&As[r * ASTR + kc] = out;
            }
        } else {
            for (int i = tid; i < GEMM_TM * GEMM_TK; i += 256) {
                int r = i >> 5, k = i & 31;
                int gr = row0 + r, gk = k0 + k;
                bf16_t v = (gr < Mrows && gk < Kdim) ? (bf16_t)Ag[(size_t)gr * lda + gk]
                                                     : (bf16_t)0.0f;
                As[r * ASTR + k] = v;
            }
        }

        // ---- stage B tile col-major (register 4x4 transpose) ----
        if (bFull && kFull) {
            int kq = (tid >> 5) * 4;             // 8 k-quads
            int cq = (tid & 31) * 4;             // 32 col-quads
            const float* bp = Bg + (size_t)(k0 + kq) * ldb + col0 + cq;
            float4 r0 = *(const float4*)(bp);
            float4 r1 = *(const float4*)(bp + (size_t)ldb);
            float4 r2 = *(const float4*)(bp + 2 * (size_t)ldb);
            float4 r3 = *(const float4*)(bp + 3 * (size_t)ldb);
            v4bf c0, c1, c2, c3;
            c0[0] = (bf16_t)r0.x; c0[1] = (bf16_t)r1.x; c0[2] = (bf16_t)r2.x; c0[3] = (bf16_t)r3.x;
            c1[0] = (bf16_t)r0.y; c1[1] = (bf16_t)r1.y; c1[2] = (bf16_t)r2.y; c1[3] = (bf16_t)r3.y;
            c2[0] = (bf16_t)r0.z; c2[1] = (bf16_t)r1.z; c2[2] = (bf16_t)r2.z; c2[3] = (bf16_t)r3.z;
            c3[0] = (bf16_t)r0.w; c3[1] = (bf16_t)r1.w; c3[2] = (bf16_t)r2.w; c3[3] = (bf16_t)r3.w;
            *(v4bf*)&Bs[(cq + 0) * BSTR + kq] = c0;
            *(v4bf*)&Bs[(cq + 1) * BSTR + kq] = c1;
            *(v4bf*)&Bs[(cq + 2) * BSTR + kq] = c2;
            *(v4bf*)&Bs[(cq + 3) * BSTR + kq] = c3;
        } else {
            for (int i = tid; i < GEMM_TK * GEMM_TN; i += 256) {
                int c = i & 127, k = i >> 7;
                int gk = k0 + k, gc = col0 + c;
                bf16_t v = (gk < Kdim && gc < Ncols) ? (bf16_t)Bg[(size_t)gk * ldb + gc]
                                                     : (bf16_t)0.0f;
                Bs[c * BSTR + k] = v;
            }
        }

        // prefetch next K-tile while we compute this one
        if (k0 + GEMM_TK < Kdim) {
            int pr = row0 + (tid >> 1);
            if (pr < Mrows)
                __builtin_prefetch(&Ag[(size_t)pr * lda + k0 + GEMM_TK + (tid & 1) * 16], 0, 1);
            int pk = k0 + GEMM_TK + (tid >> 7) * 16;
            int pc = col0 + (tid & 127);
            if (pc < Ncols)
                __builtin_prefetch(&Bg[(size_t)pk * ldb + pc], 0, 1);
        }
        __syncthreads();

        // ---- fragments: all-contiguous 16B LDS loads (ISA 7.12.2 layouts) ----
        frag16 afrag[2], bfrag[4];
        #pragma unroll
        for (int i = 0; i < 2; ++i) {
            const bf16_t* ap = &As[(wr * 32 + i * 16 + lm) * ASTR];
            afrag[i].h[0] = *(const v8bf*)(ap + lg * 8);        // K = lg*8 + 0..7
            afrag[i].h[1] = *(const v8bf*)(ap + 16 + lg * 8);   // K = 16 + lg*8 + 0..7
        }
        #pragma unroll
        for (int j = 0; j < 4; ++j) {
            const bf16_t* bp = &Bs[(wc * 64 + j * 16 + lm) * BSTR + lg * 8];
            bfrag[j].h[0] = *(const v8bf*)(bp);                 // K = lg*8 + 0..7
            bfrag[j].h[1] = *(const v8bf*)(bp + 16);            // K = 16 + lg*8 + 0..7
        }

        #pragma unroll
        for (int i = 0; i < 2; ++i)
            #pragma unroll
            for (int j = 0; j < 4; ++j)
                acc[i][j] = __builtin_amdgcn_wmma_f32_16x16x32_bf16(
                    false, afrag[i].v, false, bfrag[j].v, (short)0, acc[i][j],
                    false, false);
        __syncthreads();
    }

    // ---- epilogue: D layout -> col n = lane&15 (+group), vgpr e: m = lg*8+e ----
    #pragma unroll
    for (int j = 0; j < 4; ++j) {
        int gc = col0 + wc * 64 + j * 16 + lm;
        if (gc >= Ncols) continue;
        float bv = bias ? bias[gc] : 0.0f;
        #pragma unroll
        for (int i = 0; i < 2; ++i) {
            int base = row0 + wr * 32 + i * 16 + lg * 8;
            #pragma unroll
            for (int e = 0; e < 8; ++e) {
                int gr = base + e;
                if (gr < Mrows) {
                    size_t idx = (size_t)gr * ldc + gc;
                    float val = acc[i][j][e] + bv;
                    if (accumulate) Cg[idx] += val; else Cg[idx] = val;
                }
            }
        }
    }
}

// ----------------------------- helper kernels ------------------------------

__global__ void aqd_f32_to_bf16(const float* __restrict__ in, bf16_t* __restrict__ out,
                                size_t n)
{
    size_t i = (size_t)blockIdx.x * blockDim.x + threadIdx.x;
    if (i < n) out[i] = (bf16_t)in[i];
}

// out[cols, rows] (bf16) = in[rows, cols]^T (fp32)
__global__ void aqd_transpose_bf16(const float* __restrict__ in, bf16_t* __restrict__ out,
                                   int rows, int cols)
{
    __shared__ float tile[32][33];
    int bx = blockIdx.x * 32;   // col tile of input
    int by = blockIdx.y * 32;   // row tile of input
    int tx = threadIdx.x, ty = threadIdx.y;   // 32 x 8
    for (int i = 0; i < 32; i += 8) {
        int r = by + ty + i, c = bx + tx;
        if (r < rows && c < cols) tile[ty + i][tx] = in[(size_t)r * cols + c];
    }
    __syncthreads();
    for (int i = 0; i < 32; i += 8) {
        int r = bx + ty + i, c = by + tx;
        if (r < cols && c < rows) out[(size_t)r * rows + c] = (bf16_t)tile[tx][ty + i];
    }
}

__global__ void aqd_zero(float* p, int n)
{
    int i = blockIdx.x * blockDim.x + threadIdx.x;
    if (i < n) p[i] = 0.0f;
}

// per-channel partial sums / sumsq; 256 threads == 256 channels (coalesced)
__global__ void aqd_bn_partial(const float* __restrict__ x, int R, int Hc,
                               float* __restrict__ sums)
{
    int c = threadIdx.x;
    if (c >= Hc) return;
    int per = (R + gridDim.x - 1) / gridDim.x;
    int r0 = blockIdx.x * per;
    int r1 = r0 + per; if (r1 > R) r1 = R;
    float s = 0.0f, s2 = 0.0f;
    for (int r = r0; r < r1; ++r) {
        float v = x[(size_t)r * Hc + c];
        s += v; s2 += v * v;
    }
    atomicAdd(&sums[c], s);
    atomicAdd(&sums[Hc + c], s2);
}

__global__ void aqd_bn_finalize(float* sums, int R, int Hc)
{
    int c = blockIdx.x * blockDim.x + threadIdx.x;
    if (c < Hc) {
        float m = sums[c] / (float)R;
        float v = sums[Hc + c] / (float)R - m * m;
        sums[c] = m;
        sums[Hc + c] = rsqrtf(v + 1e-5f);
    }
}

__global__ void aqd_bn_apply(const float* __restrict__ x, float* __restrict__ y,
                             const float* __restrict__ sums,
                             const float* __restrict__ g, const float* __restrict__ b,
                             int R, int Hc, int relu)
{
    size_t i = (size_t)blockIdx.x * blockDim.x + threadIdx.x;
    size_t total = (size_t)R * Hc;
    if (i < total) {
        int c = (int)(i % Hc);
        float v = (x[i] - sums[c]) * sums[Hc + c] * g[c] + b[c];
        if (relu && v < 0.0f) v = 0.0f;
        y[i] = v;
    }
}

__global__ void aqd_sigmoid(float* x, size_t n)
{
    size_t i = (size_t)blockIdx.x * blockDim.x + threadIdx.x;
    if (i < n) x[i] = 1.0f / (1.0f + __expf(-x[i]));
}

// ----------------------------- host orchestration --------------------------

static void gemm_f(hipStream_t s, const float* A, int lda, const float* B, int ldb,
                   float* C, int ldc, int M_, int N_, int K_,
                   const float* bias, int acc)
{
    dim3 grid((N_ + GEMM_TN - 1) / GEMM_TN, (M_ + GEMM_TM - 1) / GEMM_TM);
    aqd_gemm_wmma<float><<<grid, 256, 0, s>>>(A, lda, B, ldb, C, ldc, M_, N_, K_, bias, acc);
}

static void gemm_h(hipStream_t s, const bf16_t* A, int lda, const float* B, int ldb,
                   float* C, int ldc, int M_, int N_, int K_,
                   const float* bias, int acc)
{
    dim3 grid((N_ + GEMM_TN - 1) / GEMM_TN, (M_ + GEMM_TM - 1) / GEMM_TM);
    aqd_gemm_wmma<bf16_t><<<grid, 256, 0, s>>>(A, lda, B, ldb, C, ldc, M_, N_, K_, bias, acc);
}

static void batchnorm(hipStream_t s, const float* x, float* y, int R,
                      const float* g, const float* b, float* sums, int relu)
{
    const int Hc = 256;
    aqd_zero<<<2, 256, 0, s>>>(sums, 2 * Hc);
    aqd_bn_partial<<<64, 256, 0, s>>>(x, R, Hc, sums);
    aqd_bn_finalize<<<1, 256, 0, s>>>(sums, R, Hc);
    size_t total = (size_t)R * Hc;
    aqd_bn_apply<<<(unsigned)((total + 255) / 256), 256, 0, s>>>(x, y, sums, g, b, R, Hc, relu);
}

enum {
    P_Wg1 = 0, P_bg1, P_Ws1, P_bs1, P_Wa1, P_ba1, P_Wsg1, P_bsg1, P_Wss1, P_bss1,
    P_Wsa1, P_bsa1, P_Wc1, P_bc1, P_g1, P_be1, P_gGE1, P_bGE1, P_gAE1, P_bAE1,
    P_Wg2, P_bg2, P_Ws2, P_bs2, P_Wa2, P_ba2, P_Wsg2, P_bsg2, P_Wss2, P_bss2,
    P_Wsa2, P_bsa2, P_Wc2, P_bc2, P_g2, P_be2, P_gGE2, P_bGE2, P_gAE2, P_bAE2,
    P_Wg3, P_bg3, P_Ws3, P_bs3, P_Wa3, P_ba3, P_Wsg3, P_bsg3, P_Wss3, P_bss3,
    P_Wc3, P_bc3
};

extern "C" void kernel_launch(void* const* d_in, const int* in_sizes, int n_in,
                              void* d_out, int out_size, void* d_ws, size_t ws_size,
                              hipStream_t stream)
{
    (void)in_sizes; (void)n_in; (void)out_size; (void)ws_size;

    const int Nn = 4096, Ff = 1024, Hh = 256, Cc = 128, Bb = 2, Mm = 1024;

    const float* node = (const float*)d_in[0];   // [B,N,2]
    const float* att  = (const float*)d_in[1];   // [B,M,1]
    const float* adj  = (const float*)d_in[2];   // [N,N]
    const float* Fadj = (const float*)d_in[3];   // [N,M]
    const float* feat = (const float*)d_in[4];   // [N,F]
    const float* prm[52];
    for (int i = 0; i < 52; ++i) prm[i] = (const float*)d_in[5 + i];

    // ---- workspace layout (bytes, 256B aligned chunks) ----
    char* base = (char*)d_ws;
    size_t off = 0;
    auto WSB = [&](size_t bytes) {
        void* p = base + off; off += (bytes + 255) & ~(size_t)255; return p;
    };
    bf16_t* adjh   = (bf16_t*)WSB((size_t)Nn * Nn * 2);   // bf16 adj    (L2 resident)
    bf16_t* Fadjh  = (bf16_t*)WSB((size_t)Nn * Mm * 2);   // bf16 Fadj
    bf16_t* FadjTh = (bf16_t*)WSB((size_t)Mm * Nn * 2);   // bf16 Fadj^T
    float* m1   = (float*)WSB((size_t)Nn * Hh * 4);       // layer1 graph branch (batch-shared)
    float* m1b  = (float*)WSB((size_t)Nn * Hh * 4);       // relu(bn(m1))
    float* g1a  = (float*)WSB((size_t)Nn * Hh * 4);       // layer2 model1 (batch-shared)
    float* g1b  = (float*)WSB((size_t)Nn * Hh * 4);       // relu(bn(g1a))
    float* tNH  = (float*)WSB((size_t)Nn * Hh * 4);       // temp [N,H] (also [N,C])
    float* tMH  = (float*)WSB((size_t)Mm * Hh * 4);       // temp [M,H] (also [M,C])
    float* m2   = (float*)WSB((size_t)Bb * Nn * Hh * 4);  // layer1 m2 / layer2 model2
    float* m3b  = (float*)WSB((size_t)Bb * Nn * Hh * 4);  // layer1 m3 / layer2 model3
    float* mdlA = (float*)WSB((size_t)Bb * Nn * Hh * 4);  // layer1 "model"
    float* mdlB = (float*)WSB((size_t)Bb * Nn * Hh * 4);  // layer2 "model"
    float* m3m  = (float*)WSB((size_t)Bb * Mm * Hh * 4);
    float* mAE  = (float*)WSB((size_t)Bb * Mm * Hh * 4);  // model_AE
    float* m3m2 = (float*)WSB((size_t)Bb * Mm * Hh * 4);
    float* p13  = (float*)WSB((size_t)Nn * Cc * 4);       // layer3 model1 (batch-shared)
    float* p23  = (float*)WSB((size_t)Bb * Nn * Cc * 4);
    float* p33  = (float*)WSB((size_t)Bb * Nn * Cc * 4);
    float* sums = (float*)WSB(2 * 256 * 4);

    // ---- one-time operand conversions ----
    {
        size_t na = (size_t)Nn * Nn;
        aqd_f32_to_bf16<<<(unsigned)((na + 255) / 256), 256, 0, stream>>>(adj, adjh, na);
        size_t nf = (size_t)Nn * Mm;
        aqd_f32_to_bf16<<<(unsigned)((nf + 255) / 256), 256, 0, stream>>>(Fadj, Fadjh, nf);
        dim3 tb(32, 8), tg((Mm + 31) / 32, (Nn + 31) / 32);
        aqd_transpose_bf16<<<tg, tb, 0, stream>>>(Fadj, FadjTh, Nn, Mm);
    }

    // ================= layer 1 =================
    gemm_f(stream, feat, Ff, prm[P_Wg1], Hh, tNH, Hh, Nn, Hh, Ff, nullptr, 0);
    gemm_h(stream, adjh, Nn, tNH, Hh, m1, Hh, Nn, Hh, Nn, prm[P_bg1], 0);
    gemm_f(stream, feat, Ff, prm[P_Wsg1], Hh, m1, Hh, Nn, Hh, Ff, prm[P_bsg1], 1);

    for (int b = 0; b < Bb; ++b) {
        size_t bo = (size_t)b * Nn * Hh;
        const float* nodeb = node + (size_t)b * Nn * 2;
        const float* attb  = att  + (size_t)b * Mm;
        // m2 = adj @ (node@Ws1) + bs1 + node@Wss1 + bss1
        gemm_f(stream, nodeb, 2, prm[P_Ws1], Hh, tNH, Hh, Nn, Hh, 2, nullptr, 0);
        gemm_h(stream, adjh, Nn, tNH, Hh, m2 + bo, Hh, Nn, Hh, Nn, prm[P_bs1], 0);
        gemm_f(stream, nodeb, 2, prm[P_Wss1], Hh, m2 + bo, Hh, Nn, Hh, 2, prm[P_bss1], 1);
        // m3 = Fadj @ (att@Wa1) + ba1
        gemm_f(stream, attb, 1, prm[P_Wa1], Hh, tMH, Hh, Mm, Hh, 1, nullptr, 0);
        gemm_h(stream, Fadjh, Mm, tMH, Hh, m3b + bo, Hh, Nn, Hh, Mm, prm[P_ba1], 0);
        // model = [m1 | m2 | m3] @ Wc1 + bc1 (Wc1 split by row blocks)
        gemm_f(stream, m1,       Hh, prm[P_Wc1] + 0 * Hh * Hh, Hh, mdlA + bo, Hh, Nn, Hh, Hh, prm[P_bc1], 0);
        gemm_f(stream, m2 + bo,  Hh, prm[P_Wc1] + 1 * Hh * Hh, Hh, mdlA + bo, Hh, Nn, Hh, Hh, nullptr, 1);
        gemm_f(stream, m3b + bo, Hh, prm[P_Wc1] + 2 * Hh * Hh, Hh, mdlA + bo, Hh, Nn, Hh, Hh, nullptr, 1);
    }
    batchnorm(stream, mdlA, mdlA, Bb * Nn, prm[P_g1], prm[P_be1], sums, 1);
    batchnorm(stream, m1, m1b, Nn, prm[P_gGE1], prm[P_bGE1], sums, 1);  // broadcast => stats over N

    // ================= layer 2 =================
    gemm_f(stream, m1b, Hh, prm[P_Wg2], Hh, tNH, Hh, Nn, Hh, Hh, nullptr, 0);
    gemm_h(stream, adjh, Nn, tNH, Hh, g1a, Hh, Nn, Hh, Nn, prm[P_bg2], 0);
    gemm_f(stream, m1b, Hh, prm[P_Wsg2], Hh, g1a, Hh, Nn, Hh, Hh, prm[P_bsg2], 1);

    for (int b = 0; b < Bb; ++b) {
        size_t bo = (size_t)b * Nn * Hh, bmo = (size_t)b * Mm * Hh;
        const float* attb = att + (size_t)b * Mm;
        gemm_f(stream, mdlA + bo, Hh, prm[P_Ws2], Hh, tNH, Hh, Nn, Hh, Hh, nullptr, 0);
        gemm_h(stream, adjh, Nn, tNH, Hh, m2 + bo, Hh, Nn, Hh, Nn, prm[P_bs2], 0);
        gemm_f(stream, mdlA + bo, Hh, prm[P_Wss2], Hh, m2 + bo, Hh, Nn, Hh, Hh, prm[P_bss2], 1);
        // m3m = Fadj^T @ model + att@Wsa1 + bsa1
        gemm_f(stream, attb, 1, prm[P_Wsa1], Hh, m3m + bmo, Hh, Mm, Hh, 1, prm[P_bsa1], 0);
        gemm_h(stream, FadjTh, Nn, mdlA + bo, Hh, m3m + bmo, Hh, Mm, Hh, Nn, nullptr, 1);
    }
    batchnorm(stream, m3m, mAE, Bb * Mm, prm[P_gAE1], prm[P_bAE1], sums, 1);

    for (int b = 0; b < Bb; ++b) {
        size_t bo = (size_t)b * Nn * Hh, bmo = (size_t)b * Mm * Hh;
        gemm_f(stream, mAE + bmo, Hh, prm[P_Wa2], Hh, tMH, Hh, Mm, Hh, Hh, nullptr, 0);
        gemm_h(stream, Fadjh, Mm, tMH, Hh, m3b + bo, Hh, Nn, Hh, Mm, prm[P_ba2], 0);
        // model = [model1 | model2 | model3] @ Wc2 + bc2
        gemm_f(stream, g1a,      Hh, prm[P_Wc2] + 0 * Hh * Hh, Hh, mdlB + bo, Hh, Nn, Hh, Hh, prm[P_bc2], 0);
        gemm_f(stream, m2 + bo,  Hh, prm[P_Wc2] + 1 * Hh * Hh, Hh, mdlB + bo, Hh, Nn, Hh, Hh, nullptr, 1);
        gemm_f(stream, m3b + bo, Hh, prm[P_Wc2] + 2 * Hh * Hh, Hh, mdlB + bo, Hh, Nn, Hh, Hh, nullptr, 1);
    }
    batchnorm(stream, mdlB, mdlB, Bb * Nn, prm[P_g2], prm[P_be2], sums, 1);
    batchnorm(stream, g1a, g1b, Nn, prm[P_gGE2], prm[P_bGE2], sums, 1);

    // ================= layer 3 =================
    gemm_f(stream, g1b, Hh, prm[P_Wg3], Cc, tNH, Cc, Nn, Cc, Hh, nullptr, 0);
    gemm_h(stream, adjh, Nn, tNH, Cc, p13, Cc, Nn, Cc, Nn, prm[P_bg3], 0);
    gemm_f(stream, g1b, Hh, prm[P_Wsg3], Cc, p13, Cc, Nn, Cc, Hh, prm[P_bsg3], 1);

    for (int b = 0; b < Bb; ++b) {
        size_t bo = (size_t)b * Nn * Hh, bco = (size_t)b * Nn * Cc, bmo = (size_t)b * Mm * Hh;
        gemm_f(stream, mdlB + bo, Hh, prm[P_Ws3], Cc, tNH, Cc, Nn, Cc, Hh, nullptr, 0);
        gemm_h(stream, adjh, Nn, tNH, Cc, p23 + bco, Cc, Nn, Cc, Nn, prm[P_bs3], 0);
        gemm_f(stream, mdlB + bo, Hh, prm[P_Wss3], Cc, p23 + bco, Cc, Nn, Cc, Hh, prm[P_bss3], 1);
        // m3m2 = Fadj^T @ model + model_AE@Wsa2 + bsa2
        gemm_f(stream, mAE + bmo, Hh, prm[P_Wsa2], Hh, m3m2 + bmo, Hh, Mm, Hh, Hh, prm[P_bsa2], 0);
        gemm_h(stream, FadjTh, Nn, mdlB + bo, Hh, m3m2 + bmo, Hh, Mm, Hh, Nn, nullptr, 1);
    }
    batchnorm(stream, m3m2, m3m2, Bb * Mm, prm[P_gAE2], prm[P_bAE2], sums, 1);

    float* outp = (float*)d_out;
    for (int b = 0; b < Bb; ++b) {
        size_t bco = (size_t)b * Nn * Cc, bmo = (size_t)b * Mm * Hh;
        gemm_f(stream, m3m2 + bmo, Hh, prm[P_Wa3], Cc, tMH, Cc, Mm, Cc, Hh, nullptr, 0);
        gemm_h(stream, Fadjh, Mm, tMH, Cc, p33 + bco, Cc, Nn, Cc, Mm, prm[P_ba3], 0);
        // out = [model1 | model2 | model3] @ Wc3 + bc3
        gemm_f(stream, p13,       Cc, prm[P_Wc3] + 0 * Cc * Cc, Cc, outp + bco, Cc, Nn, Cc, Cc, prm[P_bc3], 0);
        gemm_f(stream, p23 + bco, Cc, prm[P_Wc3] + 1 * Cc * Cc, Cc, outp + bco, Cc, Nn, Cc, Cc, nullptr, 1);
        gemm_f(stream, p33 + bco, Cc, prm[P_Wc3] + 2 * Cc * Cc, Cc, outp + bco, Cc, Nn, Cc, Cc, nullptr, 1);
    }
    size_t total = (size_t)Bb * Nn * Cc;
    aqd_sigmoid<<<(unsigned)((total + 255) / 256), 256, 0, stream>>>(outp, total);
}
